// MultiHeadAttentionLayer_20495583936724
// MI455X (gfx1250) — matrix operand order repaired and checked by reading, not compiled, not run on hardware
//
#include <hip/hip_runtime.h>
#include <hip/hip_bf16.h>

typedef _Float16 half_t;
typedef __attribute__((ext_vector_type(16))) _Float16 v16h;
typedef __attribute__((ext_vector_type(8)))  float    v8f;

#define D_MODEL   1024
#define NUM_HEADS 16
#define HEAD_DIM  64
#define BATCH     2
#define SEQ       2048
#define NROWS     (BATCH * SEQ)
#define NEG_BIG   (-1e9f)

// ---------------------------------------------------------------------------
// WMMA fragment loaders (wave32, 16x16x32 f16 shapes, contiguous k order)
// ---------------------------------------------------------------------------

// A fragment: 16x32 (MxK), row-major source, one contiguous 32B read/lane (k = 16*hi + e).
__device__ __forceinline__ v16h load_a_frag(const half_t* __restrict__ base,
                                            int ld, int row0, int k0) {
    int lane = threadIdx.x;
    int m = lane & 15;
    int hi = lane >> 4;
    const half_t* p = base + (size_t)(row0 + m) * ld + k0 + hi * 16;
    v16h a;
#pragma unroll
    for (int e = 0; e < 16; ++e) {
        int k = e;
        a[e] = p[k];
    }
    return a;
}

__device__ __forceinline__ v8f wmma_f16(v16h a, v16h b, v8f c) {
    return __builtin_amdgcn_wmma_f32_16x16x32_f16(
        /*neg_a=*/false, a, /*neg_b=*/false, b,
        /*c_mod=*/(short)0, c, /*reuse_a=*/false, /*reuse_b=*/false);
}

// ---------------------------------------------------------------------------
// Elementwise f32 -> f16
// ---------------------------------------------------------------------------
__global__ void cvt_f32_to_f16(const float* __restrict__ x,
                               half_t* __restrict__ y, int n) {
    int i = blockIdx.x * blockDim.x + threadIdx.x;
    if (i < n) y[i] = (half_t)x[i];
}

// ---------------------------------------------------------------------------
// Transposing f32 -> f16: wt[n*K + k] = (half)w[k*N + n]  (LDS tile transpose)
// ---------------------------------------------------------------------------
__global__ __launch_bounds__(256)
void transpose_cvt(const float* __restrict__ w, half_t* __restrict__ wt,
                   int K, int N) {
    __shared__ float tile[32][33];
    int nb = blockIdx.x * 32;
    int kb = blockIdx.y * 32;
    int tx = threadIdx.x, ty = threadIdx.y;   // 32 x 8
#pragma unroll
    for (int j = 0; j < 32; j += 8)
        tile[ty + j][tx] = w[(size_t)(kb + ty + j) * N + nb + tx];
    __syncthreads();
#pragma unroll
    for (int j = 0; j < 32; j += 8)
        wt[(size_t)(nb + ty + j) * K + kb + tx] = (half_t)tile[tx][ty + j];
}

// ---------------------------------------------------------------------------
// Tiled WMMA GEMM with block-shared, async-staged B panels (ASYNCcnt path).
// C[M,N] = A[M,K] @ W[K,N] + bias; W supplied TRANSPOSED (Wt[N,K]).
// ---------------------------------------------------------------------------
__global__ __launch_bounds__(128)
void gemm_wmma(const half_t* __restrict__ A, const half_t* __restrict__ Wt,
               const float* __restrict__ bias,
               half_t* __restrict__ Ch, float* __restrict__ Cf,
               half_t* __restrict__ Cvt,
               int M, int K, int N, float scale) {
    constexpr int BROW = 48;  // 32 data halves + 16 pad (96B rows, 32B-aligned)
    __shared__ __align__(64) half_t bstage[2][64][BROW];

    int tid  = threadIdx.y * 32 + threadIdx.x;        // 0..127
    int rt   = blockIdx.x * blockDim.y + threadIdx.y; // 16-row tile index
    int col0 = blockIdx.y * 64;                       // 64-col block
    int row0 = rt * 16;
    int lane = threadIdx.x;
    int n = lane & 15, hi = lane >> 4;

    // staging assignment: thread t covers col = t>>1, k-halves [(t&1)*16, +16)
    int scol  = tid >> 1;
    int skoff = (tid & 1) * 16;
    const half_t* wsrc = Wt + (size_t)(col0 + scol) * K + skoff;
    unsigned lds0 = (unsigned)(size_t)&bstage[0][scol][skoff];
    unsigned lds1 = (unsigned)(size_t)&bstage[1][scol][skoff];

    // IOFFSET applies to BOTH the LDS and global address (ISA 08 §4.4).
    auto stage = [&](unsigned ldsdst, int k0) {
        unsigned long long g = (unsigned long long)(wsrc + k0);
        asm volatile(
            "global_load_async_to_lds_b128 %0, %1, off\n\t"
            "global_load_async_to_lds_b128 %0, %1, off offset:16"
            :: "v"(ldsdst), "v"(g) : "memory");
    };

    v8f acc[4] = {{}, {}, {}, {}};

    auto compute = [&](int buf, v16h a) {
#pragma unroll
        for (int j = 0; j < 4; ++j) {
            v16h bf = *(const v16h*)&bstage[buf][j * 16 + n][hi * 16];
            acc[j] = wmma_f16(a, bf, acc[j]);
        }
    };

    stage(lds0, 0);
    stage(lds1, 32);
    v16h a0 = load_a_frag(A, K, row0, 0);
    v16h a1 = load_a_frag(A, K, row0, 32);

    for (int k0 = 0; k0 < K; k0 += 64) {
        bool more = (k0 + 64) < K;

        asm volatile("s_wait_asynccnt 0x2" ::: "memory");  // panel0 landed
        __syncthreads();
        compute(0, a0);
        a0 = load_a_frag(A, K, row0, more ? k0 + 64 : 0);
        __syncthreads();
        if (more) stage(lds0, k0 + 64);

        if (more) { asm volatile("s_wait_asynccnt 0x2" ::: "memory"); }
        else      { asm volatile("s_wait_asynccnt 0x0" ::: "memory"); }
        __syncthreads();
        compute(1, a1);
        a1 = load_a_frag(A, K, row0, more ? k0 + 96 : 32);
        __syncthreads();
        if (more) stage(lds1, k0 + 96);
    }

#pragma unroll
    for (int j = 0; j < 4; ++j) {
        int col = col0 + j * 16 + n;
        float bcol = bias ? bias[col] : 0.0f;
#pragma unroll
        for (int r = 0; r < 8; ++r) {
            int row = row0 + r + hi * 8;
            float v = (acc[j][r] + bcol) * scale;
            if (Cvt) {
                int bb = row >> 11;            // row / SEQ
                int s  = row & (SEQ - 1);
                int h  = col >> 6;             // col / HEAD_DIM
                int d  = col & (HEAD_DIM - 1);
                Cvt[((size_t)(bb * NUM_HEADS + h) * HEAD_DIM + d) * SEQ + s] =
                    (half_t)v;
            } else if (Ch) {
                Ch[(size_t)row * N + col] = (half_t)v;
            } else {
                Cf[(size_t)row * N + col] = v;
            }
        }
    }
}

// ---------------------------------------------------------------------------
// Flash-attention with block-shared, async-staged K/V panels.
// All 4 waves of a block share (b,h): the 32-key K panel (32x64, row-major)
// and V panel (64x32, head-transposed) are staged into LDS ONCE per block
// via GLOBAL_LOAD_ASYNC_TO_LDS_B128, double-buffered. Score and PV
// B-fragments become aligned 32B v16h LDS reads. P still goes through a
// wave-private LDS bounce in A-fragment order.
// ---------------------------------------------------------------------------
__global__ __launch_bounds__(128)
void attn_wmma(const half_t* __restrict__ Qh, const half_t* __restrict__ Kh,
               const half_t* __restrict__ Vt, const int* __restrict__ mask,
               half_t* __restrict__ Ctx) {
    constexpr int KROW = 80;  // 64 data + 16 pad halves (160B rows, 32B-align)
    constexpr int VROW = 48;  // 32 data + 16 pad halves (96B rows, 32B-align)
    __shared__ __align__(64) half_t kstage[2][32][KROW];
    __shared__ __align__(64) half_t vstage[2][64][VROW];
    __shared__ __align__(32) half_t plds[4][32][16];

    int w    = threadIdx.y;
    int lane = threadIdx.x;
    int tid  = w * 32 + lane;
    int n  = lane & 15;
    int hi = lane >> 4;

    int qt = blockIdx.x * blockDim.y + w;
    int h  = blockIdx.y;
    int b  = blockIdx.z;
    int q0 = qt * 16;

    const half_t* Qb = Qh + (size_t)b * SEQ * D_MODEL + h * HEAD_DIM;
    const half_t* Kb = Kh + (size_t)b * SEQ * D_MODEL + h * HEAD_DIM;
    const half_t* Vb = Vt + (size_t)(b * NUM_HEADS + h) * HEAD_DIM * SEQ;
    const int*    Mb = mask + (size_t)b * SEQ * SEQ;

    // staging assignment:
    //  K panel: 32 keys x 128B -> thread t covers key t>>2, 32B chunk (t&3)
    //  V panel: 64 dims x 64B  -> thread t covers dim t>>1, 32B chunk (t&1)
    int krow = tid >> 2, kch = (tid & 3) * 16;
    int vrow = tid >> 1, vch = (tid & 1) * 16;
    const half_t* ksrc = Kb + (size_t)krow * D_MODEL + kch;
    const half_t* vsrc = Vb + (size_t)vrow * SEQ + vch;
    unsigned klds[2] = {(unsigned)(size_t)&kstage[0][krow][kch],
                        (unsigned)(size_t)&kstage[1][krow][kch]};
    unsigned vlds[2] = {(unsigned)(size_t)&vstage[0][vrow][vch],
                        (unsigned)(size_t)&vstage[1][vrow][vch]};

    // 4 async ops per wave per chunk (2 K + 2 V), ASYNCcnt-tracked in order
    auto stageKV = [&](int buf, int sk0) {
        unsigned long long kg = (unsigned long long)(ksrc + (size_t)sk0 * D_MODEL);
        unsigned long long vg = (unsigned long long)(vsrc + sk0);
        asm volatile(
            "global_load_async_to_lds_b128 %0, %2, off\n\t"
            "global_load_async_to_lds_b128 %0, %2, off offset:16\n\t"
            "global_load_async_to_lds_b128 %1, %3, off\n\t"
            "global_load_async_to_lds_b128 %1, %3, off offset:16"
            :: "v"(klds[buf]), "v"(vlds[buf]), "v"(kg), "v"(vg) : "memory");
    };

    // Q tile fragments (16x64 -> two K-chunks of 32), reused for all keys
    v16h qa0 = load_a_frag(Qb, D_MODEL, q0, 0);
    v16h qa1 = load_a_frag(Qb, D_MODEL, q0, 32);

    v8f o0 = {}, o1 = {}, o2 = {}, o3 = {};   // 16x64 output accumulator
    float mrow[8], lrow[8];
#pragma unroll
    for (int r = 0; r < 8; ++r) { mrow[r] = -1e30f; lrow[r] = 0.0f; }

    // softmax + PV for one 32-key chunk; V fragments read from vstage[buf]
    auto softmax_pv = [&](int sk0, v8f s0, v8f s1, int buf) {
#pragma unroll
        for (int r = 0; r < 8; ++r) {
            int q = q0 + r + hi * 8;
            const int* mq = Mb + (size_t)q * SEQ + sk0 + n;
            float x0 = (mq[0]  != 0) ? s0[r] : NEG_BIG;
            float x1 = (mq[16] != 0) ? s1[r] : NEG_BIG;

            float mx = fmaxf(x0, x1);
#pragma unroll
            for (int d = 1; d < 16; d <<= 1)
                mx = fmaxf(mx, __shfl_xor(mx, d, 32));   // 16-lane column group

            float mnew  = fmaxf(mrow[r], mx);
            float alpha = __expf(mrow[r] - mnew);
            float p0    = __expf(x0 - mnew);
            float p1    = __expf(x1 - mnew);
            float ps    = p0 + p1;
#pragma unroll
            for (int d = 1; d < 16; d <<= 1)
                ps += __shfl_xor(ps, d, 32);

            lrow[r] = lrow[r] * alpha + ps;
            mrow[r] = mnew;
            o0[r] *= alpha; o1[r] *= alpha; o2[r] *= alpha; o3[r] *= alpha;

            // scatter P directly into A-fragment order (k = 16*hi + s):
            // p0 has colv=n -> lane row r+8hi, p1 has colv=16+n -> lane row +16.
            int Lrow = (r + hi * 8);
            plds[w][Lrow][n]      = (half_t)p0;
            plds[w][Lrow + 16][n] = (half_t)p1;
        }
        // Same-wave LDS ops are in-order (DScnt); buffer is wave-private.
        v16h pa = *(const v16h*)&plds[w][lane][0];
        o0 = wmma_f16(pa, *(const v16h*)&vstage[buf][0  + n][hi * 16], o0);
        o1 = wmma_f16(pa, *(const v16h*)&vstage[buf][16 + n][hi * 16], o1);
        o2 = wmma_f16(pa, *(const v16h*)&vstage[buf][32 + n][hi * 16], o2);
        o3 = wmma_f16(pa, *(const v16h*)&vstage[buf][48 + n][hi * 16], o3);
    };

    // prologue: both K/V panels in flight
    stageKV(0, 0);
    stageKV(1, 32);

    const int nc = SEQ / 32;
    for (int c = 0; c < nc; ++c) {
        int buf = c & 1;
        int sk0 = c * 32;
        bool more = (c + 2) < nc;

        if (more) { asm volatile("s_wait_asynccnt 0x4" ::: "memory"); }
        else      { asm volatile("s_wait_asynccnt 0x0" ::: "memory"); }
        __syncthreads();                       // panel visible to whole block

        // ---- scores: S[16x32] = Qtile @ K^T (4 wmma, B-frags from LDS) ----
        v8f s0 = {}, s1 = {};
        s0 = wmma_f16(qa0, *(const v16h*)&kstage[buf][n][0  + hi * 16], s0);
        s0 = wmma_f16(qa1, *(const v16h*)&kstage[buf][n][32 + hi * 16], s0);
        s1 = wmma_f16(qa0, *(const v16h*)&kstage[buf][16 + n][0  + hi * 16], s1);
        s1 = wmma_f16(qa1, *(const v16h*)&kstage[buf][16 + n][32 + hi * 16], s1);

        softmax_pv(sk0, s0, s1, buf);

        __syncthreads();                       // all readers done with `buf`
        if (more) stageKV(buf, sk0 + 64);      // restage behind the barrier
    }

    // ---- normalize and store context (heads recombined, row-major) ----
    half_t* Cb = Ctx + (size_t)b * SEQ * D_MODEL + h * HEAD_DIM;
#pragma unroll
    for (int r = 0; r < 8; ++r) {
        int row = q0 + r + hi * 8;
        float inv = 1.0f / lrow[r];
        Cb[(size_t)row * D_MODEL + 0  + n] = (half_t)(o0[r] * inv);
        Cb[(size_t)row * D_MODEL + 16 + n] = (half_t)(o1[r] * inv);
        Cb[(size_t)row * D_MODEL + 32 + n] = (half_t)(o2[r] * inv);
        Cb[(size_t)row * D_MODEL + 48 + n] = (half_t)(o3[r] * inv);
    }
}

// ---------------------------------------------------------------------------
// Launch
// ---------------------------------------------------------------------------
extern "C" void kernel_launch(void* const* d_in, const int* in_sizes, int n_in,
                              void* d_out, int out_size, void* d_ws, size_t ws_size,
                              hipStream_t stream) {
    const float* query = (const float*)d_in[0];
    const float* key   = (const float*)d_in[1];
    const float* value = (const float*)d_in[2];
    const int*   mask  = (const int*)d_in[3];
    const float* Wq = (const float*)d_in[4];
    const float* bq = (const float*)d_in[5];
    const float* Wk = (const float*)d_in[6];
    const float* bk = (const float*)d_in[7];
    const float* Wv = (const float*)d_in[8];
    const float* bv = (const float*)d_in[9];
    const float* Wo = (const float*)d_in[10];
    const float* bo = (const float*)d_in[11];

    const size_t nio = (size_t)NROWS * D_MODEL;    // 4M elems
    const size_t nw  = (size_t)D_MODEL * D_MODEL;  // 1M elems

    char* ws = (char*)d_ws;
    half_t* xq  = (half_t*)ws; ws += nio * sizeof(half_t);
    half_t* xk  = (half_t*)ws; ws += nio * sizeof(half_t);
    half_t* xv  = (half_t*)ws; ws += nio * sizeof(half_t);
    half_t* wqt = (half_t*)ws; ws += nw  * sizeof(half_t);
    half_t* wkt = (half_t*)ws; ws += nw  * sizeof(half_t);
    half_t* wvt = (half_t*)ws; ws += nw  * sizeof(half_t);
    half_t* wot = (half_t*)ws; ws += nw  * sizeof(half_t);
    half_t* Qh  = (half_t*)ws; ws += nio * sizeof(half_t);
    half_t* Kh  = (half_t*)ws; ws += nio * sizeof(half_t);
    half_t* Vt  = (half_t*)ws; ws += nio * sizeof(half_t);
    half_t* Ctx = (half_t*)ws; ws += nio * sizeof(half_t);

    // 1) convert activations; transpose+convert weights
    {
        int t = 256;
        cvt_f32_to_f16<<<(int)((nio + t - 1) / t), t, 0, stream>>>(query, xq, (int)nio);
        cvt_f32_to_f16<<<(int)((nio + t - 1) / t), t, 0, stream>>>(key,   xk, (int)nio);
        cvt_f32_to_f16<<<(int)((nio + t - 1) / t), t, 0, stream>>>(value, xv, (int)nio);

        dim3 tg(D_MODEL / 32, D_MODEL / 32);
        dim3 tb(32, 8);
        transpose_cvt<<<tg, tb, 0, stream>>>(Wq, wqt, D_MODEL, D_MODEL);
        transpose_cvt<<<tg, tb, 0, stream>>>(Wk, wkt, D_MODEL, D_MODEL);
        transpose_cvt<<<tg, tb, 0, stream>>>(Wv, wvt, D_MODEL, D_MODEL);
        transpose_cvt<<<tg, tb, 0, stream>>>(Wo, wot, D_MODEL, D_MODEL);
    }

    // 2) QKV projections (Q carries 1/sqrt(HEAD_DIM); V stored head-transposed)
    {
        dim3 grid(NROWS / 64, D_MODEL / 64);
        dim3 block(32, 4);
        const float qscale = 0.125f;   // 1/sqrt(64)
        gemm_wmma<<<grid, block, 0, stream>>>(xq, wqt, bq, Qh, nullptr, nullptr,
                                              NROWS, D_MODEL, D_MODEL, qscale);
        gemm_wmma<<<grid, block, 0, stream>>>(xk, wkt, bk, Kh, nullptr, nullptr,
                                              NROWS, D_MODEL, D_MODEL, 1.0f);
        gemm_wmma<<<grid, block, 0, stream>>>(xv, wvt, bv, nullptr, nullptr, Vt,
                                              NROWS, D_MODEL, D_MODEL, 1.0f);
    }

    // 3) flash attention
    {
        dim3 grid(SEQ / 16 / 4, NUM_HEADS, BATCH);
        dim3 block(32, 4);
        attn_wmma<<<grid, block, 0, stream>>>(Qh, Kh, Vt, mask, Ctx);
    }

    // 4) output projection -> f32 d_out
    {
        dim3 grid(NROWS / 64, D_MODEL / 64);
        dim3 block(32, 4);
        gemm_wmma<<<grid, block, 0, stream>>>(Ctx, wot, bo, nullptr, (float*)d_out,
                                              nullptr, NROWS, D_MODEL, D_MODEL, 1.0f);
    }
}
